// YoloXLoss_77962246357519
// MI455X (gfx1250) — compile-verified, matrix operand order checked by reading, not run on hardware
//
#include <hip/hip_runtime.h>
#include <hip/hip_bf16.h>

// ---------------------------------------------------------------------------
// YOLOX SimOTA loss for MI455X (gfx1250, wave32).
// Class-cost gather is done as a one-hot 16x80 @ 80x16 matmul on the WMMA
// unit (v_wmma_f32_16x16x32_f16, f32 accumulate), exploiting G==16==WMMA M.
// All sigmoid/sqrt chains are rewritten in log-space:
//   log p = 0.5*(logsig(x)+logsig(conf)),  logsig(t)=min(t,0)-log(1+e^-|t|)
// which removes every v_div/v_sqrt sequence from the hot loop.
// Workspace requirement: ~84 MB (cost + iou matrices dominate).
// ---------------------------------------------------------------------------

#define NCLS   80
#define ATOT   33600          // 160*160 + 80*80 + 40*40
#define NB     16             // batch
#define NG     16             // ground-truth boxes per image
#define NTILES (ATOT / 16)    // 2100 sixteen-anchor tiles per image

typedef __attribute__((ext_vector_type(16))) _Float16 v16h;
typedef __attribute__((ext_vector_type(8)))  float    v8f;

// --- anchor index -> (level, local hw index, W, stride) --------------------
__device__ __forceinline__ void anchor_info(int a, int& level, int& local,
                                            int& W, float& stride) {
  if (a < 25600)      { level = 0; local = a;         W = 160; stride =  8.f; }
  else if (a < 32000) { level = 1; local = a - 25600; W =  80; stride = 16.f; }
  else                { level = 2; local = a - 32000; W =  40; stride = 32.f; }
}

__device__ __forceinline__ const float* level_ptr(const float* p0, const float* p1,
                                                  const float* p2, int level) {
  return level == 0 ? p0 : (level == 1 ? p1 : p2);
}

// logsig(x) = log(sigmoid(x)) = min(x,0) - softplus(-|x|).
// Also returns sp = log(1+exp(-|x|)) (softplus term, shared with bce(x,0)).
__device__ __forceinline__ float logsig(float x, float& sp) {
  float e = __expf(-fabsf(x));
  sp = __logf(1.f + e);
  return fminf(x, 0.f) - sp;
}

// per-class terms: bval = log1mp - logp, plus base/S0 partial sums
__device__ __forceinline__ void class_terms(float x, float lsc,
                                            float& bval, float& basep, float& s0p) {
  float sp;
  float lsx  = logsig(x, sp);
  float logp = 0.5f * (lsx + lsc);            // = log sqrt(sig(x)*sig(conf))
  float pr   = __expf(logp);                  // p in (0,1]
  float lp   = fmaxf(logp, -100.f);           // clip(log p, -100)
  float l1   = fmaxf(__logf(1.f - pr), -100.f); // clip(log(1-p), -100)
  basep     += -l1;                           // base = -sum log1mp
  s0p       += fmaxf(x, 0.f) + sp;            // bce(x, 0)
  bval       = l1 - lp;
}

// ---------------------------------------------------------------------------
// Kernel 0: init workspace (match bitmask + accumulators)
// ---------------------------------------------------------------------------
__global__ void k_init(unsigned int* match, float* accum) {
  int i = blockIdx.x * blockDim.x + threadIdx.x;
  if (i < NB * ATOT) match[i] = 0u;
  if (i < 8)         accum[i] = 0.f;
}

// ---------------------------------------------------------------------------
// Kernel 1: decode boxes + conf logits  (streams 5 of 85 channels)
// ---------------------------------------------------------------------------
__global__ void k_decode(const float* p0, const float* p1, const float* p2,
                         float4* pbox, float* pconf) {
  int i = blockIdx.x * blockDim.x + threadIdx.x;
  if (i >= NB * ATOT) return;
  int b = i / ATOT, a = i % ATOT;
  int level, local, W; float stride;
  anchor_info(a, level, local, W, stride);
  const float* p  = level_ptr(p0, p1, p2, level);
  size_t hw       = (size_t)W * W;
  const float* ch = p + (size_t)b * 85 * hw + local;
  float tx = ch[0], ty = ch[hw], tw = ch[2 * hw], th = ch[3 * hw];
  int gy = local / W, gx = local % W;
  float4 box;
  box.x = (tx + (float)gx) * stride;
  box.y = (ty + (float)gy) * stride;
  box.z = __expf(tw) * stride;
  box.w = __expf(th) * stride;
  pbox[i]  = box;
  pconf[i] = ch[4 * hw];
}

// ---------------------------------------------------------------------------
// Kernel 2: cost matrix. 8 waves/block, one 16-anchor tile per wave.
// WMMA: D(16 GT x 16 anchors) = OneHot(16x96) x Bvals(96x16), K-chunks of 32.
// VGPR layouts per CDNA5 ISA 7.12.2:
//   A 16x32 f16: lane m=L&15, elem j -> K = j + (j>=8?8:0) + (L>=16?8:0)
//   B 32x16 f16: lane n=L&15, same K packing (N across lanes, K in regs)
//   C/D 16x16 f32: lane n=L&15, VGPR r -> M = r + 8*(L>=16)
// ---------------------------------------------------------------------------
__global__ void __launch_bounds__(256)
k_cost(const float* p0, const float* p1, const float* p2, const float* labels,
       const float4* pbox, float* cost, float* iouM, float* S0) {
  __shared__ float ltb[NG][4];
  __shared__ int   ltc[NG];
  int b = blockIdx.y;
  if (threadIdx.x < NG) {
    const float* L = labels + ((size_t)b * NG + threadIdx.x) * 5;
    ltb[threadIdx.x][0] = L[0]; ltb[threadIdx.x][1] = L[1];
    ltb[threadIdx.x][2] = L[2]; ltb[threadIdx.x][3] = L[3];
    ltc[threadIdx.x] = (int)L[4];
  }
  __syncthreads();

  int wave = threadIdx.x >> 5;
  int tile = blockIdx.x * 8 + wave;
  if (tile >= NTILES) return;                  // uniform per wave, EXEC stays full

  int lane = threadIdx.x & 31;
  int hi   = lane >> 4;        // K-half selector
  int n    = lane & 15;        // anchor column (B/C/D) and GT row (A)
  int a    = tile * 16 + n;

  int level, local, W; float stride;
  anchor_info(a, level, local, W, stride);
  const float* p  = level_ptr(p0, p1, p2, level);
  size_t hw       = (size_t)W * W;
  const float* ch = p + (size_t)b * 85 * hw + local;

  float confl = ch[4 * hw];
  float spc;
  float lsc = logsig(confl, spc);              // log(sigmoid(conf))

  v8f acc = {};
  float basep = 0.f, s0p = 0.f;
  int tcm = ltc[n];                             // GT class for A-matrix row n

  #pragma unroll
  for (int kk = 0; kk < 3; ++kk) {
    int K0 = kk * 32;
    v16h amat, bmat;
    #pragma unroll
    for (int j = 0; j < 16; ++j) {
      int k = K0 + j + ((j >= 8) ? 8 : 0) + (hi ? 8 : 0);
      float bv = 0.f;
      if (k < NCLS) {
        float x = ch[(size_t)(5 + k) * hw];
        class_terms(x, lsc, bv, basep, s0p);
      }
      bmat[j] = (_Float16)bv;
      amat[j] = (_Float16)((tcm == k) ? 1.f : 0.f);
    }
    acc = __builtin_amdgcn_wmma_f32_16x16x32_f16(
        false, amat, false, bmat, (short)0, acc, false, false);
  }

  // fold the two K-halves' per-anchor sums
  basep += __shfl_xor(basep, 16, 32);
  s0p   += __shfl_xor(s0p,   16, 32);
  size_t ba = (size_t)b * ATOT + a;
  if (!hi) S0[ba] = s0p;

  // geometry per (g, anchor) pair; this lane owns rows m = r + 8*hi
  float4 pb = pbox[ba];
  int gy = local / W, gx = local % W;
  float xcf = ((float)gx + 0.5f) * stride;
  float ycf = ((float)gy + 0.5f) * stride;
  float rad = 2.5f * stride;
  float px1 = pb.x - pb.z * 0.5f, px2 = pb.x + pb.z * 0.5f;
  float py1 = pb.y - pb.w * 0.5f, py2 = pb.y + pb.w * 0.5f;
  float parea = pb.z * pb.w;

  float iouv[8];
  int interb = 0, anyb = 0;
  #pragma unroll
  for (int r = 0; r < 8; ++r) {
    int m = r + hi * 8;
    float bx = ltb[m][0], by = ltb[m][1], bw = ltb[m][2], bh = ltb[m][3];
    float bx1 = bx - bw * 0.5f, bx2 = bx + bw * 0.5f;
    float by1 = by - bh * 0.5f, by2 = by + bh * 0.5f;
    bool inb = (xcf > bx1) && (xcf < bx2) && (ycf > by1) && (ycf < by2);
    bool inm = (xcf > bx - rad) && (xcf < bx + rad) &&
               (ycf > by - rad) && (ycf < by + rad);
    if (inb && inm) interb |= (1 << r);
    if (inb || inm) anyb = 1;
    float iw = fmaxf(fminf(bx2, px2) - fmaxf(bx1, px1), 0.f);
    float ih = fmaxf(fminf(by2, py2) - fmaxf(by1, py1), 0.f);
    float inter = iw * ih;
    iouv[r] = inter / (bw * bh + parea - inter + 1e-16f);
  }
  anyb |= __shfl_xor(anyb, 16, 32);            // union over all 16 GTs
  float notu = anyb ? 0.f : 1e9f;

  #pragma unroll
  for (int r = 0; r < 8; ++r) {
    int m = r + hi * 8;
    float c = basep + acc[r]
            - 3.f * __logf(iouv[r] + 1e-8f)
            + (((interb >> r) & 1) ? 0.f : 1e5f)
            + notu;
    size_t off = ((size_t)(b * NG + m)) * ATOT + a;
    cost[off] = c;
    iouM[off] = iouv[r];
  }
}

// ---------------------------------------------------------------------------
// Kernel 3: per-GT top-10 (dyn_k from masked IoU, matching from min-cost)
// One block per (g, b). Per-thread sorted lists + log-tree LDS merge.
// ---------------------------------------------------------------------------
__global__ void __launch_bounds__(256)
k_topk(const float* cost, const float* iouM, unsigned int* match) {
  int g = blockIdx.x, b = blockIdx.y, t = threadIdx.x;
  const float* crow = cost + ((size_t)(b * NG + g)) * ATOT;
  const float* irow = iouM + ((size_t)(b * NG + g)) * ATOT;

  float ti[10], cv[10]; int ci[10];
  #pragma unroll
  for (int j = 0; j < 10; ++j) { ti[j] = -1.f; cv[j] = 3.4e38f; ci[j] = 0; }

  for (int a = t; a < ATOT; a += 256) {
    float c  = crow[a];
    float im = (c < 5e8f) ? irow[a] : 0.f;     // union <=> cost lacks BIG term
    if (im > ti[9]) {
      int j = 9;
      while (j > 0 && im > ti[j - 1]) { ti[j] = ti[j - 1]; --j; }
      ti[j] = im;
    }
    if (c < cv[9]) {
      int j = 9;
      while (j > 0 && c < cv[j - 1]) { cv[j] = cv[j - 1]; ci[j] = ci[j - 1]; --j; }
      cv[j] = c; ci[j] = a;
    }
  }

  __shared__ float s_ti[256][10];
  __shared__ float s_cv[256][10];
  __shared__ int   s_ci[256][10];
  #pragma unroll
  for (int j = 0; j < 10; ++j) { s_ti[t][j] = ti[j]; s_cv[t][j] = cv[j]; s_ci[t][j] = ci[j]; }
  __syncthreads();

  for (int off = 128; off >= 1; off >>= 1) {
    if (t < off) {
      float mi[10], mc[10]; int mii[10];
      int ia = 0, ib = 0;
      #pragma unroll
      for (int j = 0; j < 10; ++j) {
        float va = s_ti[t][ia], vb = s_ti[t + off][ib];
        if (va >= vb) { mi[j] = va; ++ia; } else { mi[j] = vb; ++ib; }
      }
      int ca = 0, cb = 0;
      #pragma unroll
      for (int j = 0; j < 10; ++j) {
        float va = s_cv[t][ca], vb = s_cv[t + off][cb];
        if (va <= vb) { mc[j] = va; mii[j] = s_ci[t][ca]; ++ca; }
        else          { mc[j] = vb; mii[j] = s_ci[t + off][cb]; ++cb; }
      }
      #pragma unroll
      for (int j = 0; j < 10; ++j) { s_ti[t][j] = mi[j]; s_cv[t][j] = mc[j]; s_ci[t][j] = mii[j]; }
    }
    __syncthreads();
  }

  if (t == 0) {
    float s = 0.f;
    #pragma unroll
    for (int j = 0; j < 10; ++j) s += s_ti[0][j];
    int dynk = (int)s;
    if (dynk < 1) dynk = 1;
    if (dynk > 10) dynk = 10;
    for (int j = 0; j < dynk; ++j)
      atomicOr(&match[(size_t)b * ATOT + s_ci[0][j]], 1u << g);
  }
}

// ---------------------------------------------------------------------------
// Kernel 4: per-anchor losses + global accumulation
// ---------------------------------------------------------------------------
__device__ __forceinline__ float wave_sum(float v) {
  #pragma unroll
  for (int o = 16; o >= 1; o >>= 1) v += __shfl_xor(v, o, 32);
  return v;
}

__global__ void __launch_bounds__(256)
k_loss(const float* p0, const float* p1, const float* p2, const float* labels,
       const float4* pbox, const float* pconf, const float* S0,
       const float* cost, const float* iouM, const unsigned int* match,
       float* accum) {
  int i = blockIdx.x * blockDim.x + threadIdx.x;
  float fg = 0.f, lbox = 0.f, lcls = 0.f, lconf = 0.f;
  if (i < NB * ATOT) {
    int b = i / ATOT, a = i % ATOT;
    unsigned mask = match[i];
    float x = pconf[i];
    if (mask) {
      fg = 1.f;
      int gidx;
      if (mask & (mask - 1)) {                 // conflict: argmin-g of full cost
        float best = 3.4e38f; gidx = 0;
        for (int g = 0; g < NG; ++g) {
          float c = cost[((size_t)(b * NG + g)) * ATOT + a];
          if (c < best) { best = c; gidx = g; }
        }
      } else {
        gidx = __ffs(mask) - 1;
      }
      float miou = iouM[((size_t)(b * NG + gidx)) * ATOT + a];
      const float* L = labels + ((size_t)(b * NG + gidx)) * 5;
      float bx = L[0], by = L[1], bw = L[2], bh = L[3];
      float4 pb = pbox[i];
      float iw = fmaxf(fminf(pb.x + pb.z * 0.5f, bx + bw * 0.5f) -
                       fmaxf(pb.x - pb.z * 0.5f, bx - bw * 0.5f), 0.f);
      float ih = fmaxf(fminf(pb.y + pb.w * 0.5f, by + bh * 0.5f) -
                       fmaxf(pb.y - pb.w * 0.5f, by - bh * 0.5f), 0.f);
      float inter = iw * ih;
      float ei = inter / (pb.z * pb.w + bw * bh - inter + 1e-16f);
      lbox = 1.f - ei * ei;
      // class loss: sum_c bce(x_c, t_c) = S0 - x_{tc} * miou
      int level, local, W; float stride;
      anchor_info(a, level, local, W, stride);
      const float* p  = level_ptr(p0, p1, p2, level);
      size_t hw       = (size_t)W * W;
      int tcg = (int)L[4];
      float xcl = p[(size_t)b * 85 * hw + (size_t)(5 + tcg) * hw + local];
      lcls = S0[i] - xcl * miou;
    }
    float sp; (void)logsig(x, sp);             // sp = log1p(exp(-|x|))
    lconf = fmaxf(x, 0.f) - x * fg + sp;       // bce(x, fg)
  }

  // block reduce 4 values -> 4 global float atomics
  float s0 = wave_sum(lbox), s1 = wave_sum(lconf), s2 = wave_sum(lcls), s3 = wave_sum(fg);
  __shared__ float red[8][4];
  int lane = threadIdx.x & 31, w = threadIdx.x >> 5;
  if (lane == 0) { red[w][0] = s0; red[w][1] = s1; red[w][2] = s2; red[w][3] = s3; }
  __syncthreads();
  if (threadIdx.x == 0) {
    float t0 = 0.f, t1 = 0.f, t2 = 0.f, t3 = 0.f;
    #pragma unroll
    for (int k = 0; k < 8; ++k) { t0 += red[k][0]; t1 += red[k][1]; t2 += red[k][2]; t3 += red[k][3]; }
    atomicAdd(accum + 0, t0);
    atomicAdd(accum + 1, t1);
    atomicAdd(accum + 2, t2);
    atomicAdd(accum + 3, t3);
  }
}

// ---------------------------------------------------------------------------
// Kernel 5: final scalar
// ---------------------------------------------------------------------------
__global__ void k_final(const float* accum, float* out) {
  out[0] = (5.f * accum[0] + accum[1] + accum[2]) / fmaxf(accum[3], 1.f);
}

// ---------------------------------------------------------------------------
extern "C" void kernel_launch(void* const* d_in, const int* in_sizes, int n_in,
                              void* d_out, int out_size, void* d_ws, size_t ws_size,
                              hipStream_t stream) {
  (void)in_sizes; (void)n_in; (void)out_size; (void)ws_size;
  const float* p0     = (const float*)d_in[0];
  const float* p1     = (const float*)d_in[1];
  const float* p2     = (const float*)d_in[2];
  const float* labels = (const float*)d_in[3];

  char* ws = (char*)d_ws;
  size_t off = 0;
  float* cost = (float*)(ws + off); off += (size_t)NB * NG * ATOT * sizeof(float);   // 34.4 MB
  float* iouM = (float*)(ws + off); off += (size_t)NB * NG * ATOT * sizeof(float);   // 34.4 MB
  float4* pbox = (float4*)(ws + off); off += (size_t)NB * ATOT * sizeof(float4);     //  8.6 MB
  float* pconf = (float*)(ws + off); off += (size_t)NB * ATOT * sizeof(float);       //  2.15 MB
  float* S0    = (float*)(ws + off); off += (size_t)NB * ATOT * sizeof(float);       //  2.15 MB
  unsigned int* match = (unsigned int*)(ws + off); off += (size_t)NB * ATOT * 4;     //  2.15 MB
  float* accum = (float*)(ws + off); off += 32;                                      // ~84 MB total

  int totBA = NB * ATOT;
  int grd   = (totBA + 255) / 256;

  k_init  <<<grd, 256, 0, stream>>>(match, accum);
  k_decode<<<grd, 256, 0, stream>>>(p0, p1, p2, pbox, pconf);
  k_cost  <<<dim3((NTILES + 7) / 8, NB), 256, 0, stream>>>(p0, p1, p2, labels,
                                                           pbox, cost, iouM, S0);
  k_topk  <<<dim3(NG, NB), 256, 0, stream>>>(cost, iouM, match);
  k_loss  <<<grd, 256, 0, stream>>>(p0, p1, p2, labels, pbox, pconf, S0,
                                    cost, iouM, match, accum);
  k_final <<<1, 1, 0, stream>>>(accum, (float*)d_out);
}
